// HarmonicMultiHeadAttention_22625887715625
// MI455X (gfx1250) — matrix operand order repaired and checked by reading, not compiled, not run on hardware
//
#include <hip/hip_runtime.h>

#define Bc 2
#define Sq 2048
#define Dm 1024
#define Hn 16
#define Hd 64

typedef __attribute__((ext_vector_type(16))) __bf16 v16bf;
typedef __attribute__((ext_vector_type(8)))  __bf16 v8bf;
typedef __attribute__((ext_vector_type(8)))  float  v8f;
typedef __attribute__((ext_vector_type(4)))  float  v4f;
typedef __attribute__((ext_vector_type(4)))  unsigned int v4u;
typedef __attribute__((ext_vector_type(8)))  int    v8i;
typedef __attribute__((ext_vector_type(4)))  int    v4i;

__device__ __forceinline__ v8f wmma_bf16(v16bf a, v16bf b, v8f c) {
    return __builtin_amdgcn_wmma_f32_16x16x32_bf16(
        false, a, false, b, (short)0, c, false, false);
}

// ---- A-fragment (16x32 MxK, 16-bit layout, ISA 7.12.2), vectorized ----
__device__ __forceinline__ v16bf load_a_bf16v(const __bf16* __restrict__ src,
                                              int ld, int r0, int k0, int lane) {
    int m = lane & 15, koff = (lane < 16) ? 0 : 8;
    const __bf16* p = src + (size_t)(r0 + m) * ld + k0;
    v8bf lo = *(const v8bf*)(p + koff);
    v8bf hi = *(const v8bf*)(p + 16 + koff);
    v16bf a;
#pragma unroll
    for (int j = 0; j < 8; ++j) { a[j] = lo[j]; a[8 + j] = hi[j]; }
    return a;
}

__device__ __forceinline__ v16bf load_a_f32v(const float* __restrict__ src,
                                             int ld, int r0, int k0, int lane) {
    int m = lane & 15, koff = (lane < 16) ? 0 : 8;
    const float* p = src + (size_t)(r0 + m) * ld + k0;
    v4f x0 = *(const v4f*)(p + koff);
    v4f x1 = *(const v4f*)(p + koff + 4);
    v4f y0 = *(const v4f*)(p + 16 + koff);
    v4f y1 = *(const v4f*)(p + 16 + koff + 4);
    v16bf a;
#pragma unroll
    for (int j = 0; j < 4; ++j) {
        a[j]      = (__bf16)x0[j];  a[4 + j]  = (__bf16)x1[j];
        a[8 + j]  = (__bf16)y0[j];  a[12 + j] = (__bf16)y1[j];
    }
    return a;
}

// ---- B-fragment (32x16 KxN): lane = N column, elements over K (table order) ----
__device__ __forceinline__ v16bf load_b_kmajor(const __bf16* __restrict__ src,
                                               size_t ld, int k0, int n0, int lane) {
    int n = lane & 15, koff = (lane < 16) ? 0 : 8;
    v16bf b;
#pragma unroll
    for (int j = 0; j < 8; ++j) {
        b[j]     = src[(size_t)(k0 + koff + j) * ld + n0 + n];
        b[8 + j] = src[(size_t)(k0 + 16 + koff + j) * ld + n0 + n];
    }
    return b;
}

// ---- TDM: stream a [rows x 64] bf16 tile (row stride 64 elems) into LDS ----
// D# per ISA 8.3/8.4: count=1, data_size=2B, tensor dims {64, Sq}, stride 64,
// tile {64, rows}, type=2.  6-arg builtin (clang-23 / therock headers);
// groups 2/3/extra zero (<=2D tensor).
__device__ __forceinline__ void tdm_load_tile64(const void* gptr, unsigned int lds_off,
                                                unsigned int rows) {
    unsigned long long ga = (unsigned long long)(uintptr_t)gptr;
    v4u g0 = { 1u,                                   // count=1 (user D#)
               lds_off,                              // lds_addr
               (unsigned int)ga,                     // global_addr[31:0]
               (unsigned int)((ga >> 32) & 0x01FFFFFFu) | (2u << 30) }; // [56:32]|type=2
    v8i g1;
    g1[0] = (1 << 16);                               // data_size = 2 bytes
    g1[1] = (int)(64u << 16);                        // tensor_dim0 = 64  (bits 79:48 lo)
    g1[2] = (int)((unsigned)Sq << 16);               // tensor_dim1 = Sq (bits 111:80 lo)
    g1[3] = (int)(64u << 16);                        // tile_dim0 = 64   (bits 127:112)
    g1[4] = (int)rows;                               // tile_dim1 = rows
    g1[5] = 64;                                      // tensor_dim0_stride = 64
    g1[6] = 0;
    g1[7] = 0;
    v4i z4 = { 0, 0, 0, 0 };
    v8i z8 = { 0, 0, 0, 0, 0, 0, 0, 0 };
    __builtin_amdgcn_tensor_load_to_lds(g0, g1, z4, z4, z8, 0);
}

// ---------------- Kernel 0: weight transpose+convert  Wt[k][n] = bf16(W[n][k])
__global__ void wt_kernel(const float* __restrict__ W, __bf16* __restrict__ Wt) {
    __shared__ float tile[32][33];
    int tx = threadIdx.x & 31, ty = threadIdx.x >> 5;
    int k0 = (blockIdx.x & 31) * 32;
    int n0 = (blockIdx.x >> 5) * 32;
#pragma unroll
    for (int i = 0; i < 4; ++i)
        tile[ty + i * 8][tx] = W[(size_t)(n0 + ty + i * 8) * Dm + k0 + tx];
    __syncthreads();
#pragma unroll
    for (int i = 0; i < 4; ++i)
        Wt[(size_t)(k0 + ty + i * 8) * Dm + n0 + tx] = (__bf16)tile[tx][ty + i * 8];
}

// ---------------- Kernel 1: projection  Y = X @ W^T + b  -> bf16 heads
template <int KMODE>
__global__ void proj_qkv_kernel(const float* __restrict__ X,
                                const __bf16* __restrict__ Wt,
                                const float* __restrict__ bias,
                                __bf16* __restrict__ dst) {
    __shared__ __bf16 Sh[8][16][64];
    const int lane = threadIdx.x & 31, wave = threadIdx.x >> 5;
    const int nchunks = Dm / 64;
    int t  = blockIdx.x * 8 + wave;
    int r0 = (t / nchunks) * 16;
    int n0 = (t % nchunks) * 64;

    v8f acc[4] = {};
    for (int k0 = 0; k0 < Dm; k0 += 32) {
        v16bf a = load_a_f32v(X, Dm, r0, k0, lane);
#pragma unroll
        for (int c = 0; c < 4; ++c) {
            v16bf b = load_b_kmajor(Wt, Dm, k0, n0 + c * 16, lane);
            acc[c] = wmma_bf16(a, b, acc[c]);
        }
    }
    int half8 = (lane >> 4) * 8, coll = lane & 15;
    int bidx = r0 >> 11;
    int h = n0 >> 6;

    if (KMODE) {
        int s0 = (r0 & (Sq - 1)) + half8;
#pragma unroll
        for (int c = 0; c < 4; ++c) {
            int d = c * 16 + coll;
            float bvv = bias[n0 + d];
            v8bf pack;
#pragma unroll
            for (int r = 0; r < 8; ++r) pack[r] = (__bf16)(acc[c][r] + bvv);
            *(v8bf*)(dst + (((size_t)bidx * Hn + h) * Hd + d) * Sq + s0) = pack;
        }
    } else {
#pragma unroll
        for (int c = 0; c < 4; ++c) {
            float bvv = bias[n0 + c * 16 + coll];
#pragma unroll
            for (int r = 0; r < 8; ++r)
                Sh[wave][r + half8][c * 16 + coll] = (__bf16)(acc[c][r] + bvv);
        }
        __builtin_amdgcn_wave_barrier();
        asm volatile("s_wait_dscnt 0x0" ::: "memory");
        int row = lane >> 1, off = (lane & 1) * 32;
        int s = (r0 + row) & (Sq - 1);
        __bf16* dp = dst + (((size_t)bidx * Hn + h) * Sq + s) * Hd + off;
        *(v16bf*)dp        = *(const v16bf*)&Sh[wave][row][off];
        *(v16bf*)(dp + 16) = *(const v16bf*)&Sh[wave][row][off + 16];
    }
}

// ---------------- Kernel 2: QK^T + modulation + online stats (software-pipelined)
__global__ void scores_kernel(const __bf16* __restrict__ qb,
                              const __bf16* __restrict__ kbT,
                              const float* __restrict__ resonance,
                              const float* __restrict__ phase,
                              float* __restrict__ attn,
                              float* __restrict__ Mrow,
                              float* __restrict__ Lrow) {
    const int lane = threadIdx.x & 31, wave = threadIdx.x >> 5;
    const int qtiles = Sq / 128;
    int bh = blockIdx.x / qtiles;
    int q0 = (blockIdx.x % qtiles) * 128 + wave * 16;

    const __bf16* qbase = qb  + (size_t)bh * Sq * Hd;
    const __bf16* kbase = kbT + (size_t)bh * Hd * Sq;
    float* arow = attn + (size_t)bh * Sq * Sq;
    int h = bh & (Hn - 1);
    float res = resonance[h], ph = phase[h];
    const float scale = 0.125f;

    v16bf a0 = load_a_bf16v(qbase, Hd, q0, 0, lane);
    v16bf a1 = load_a_bf16v(qbase, Hd, q0, 32, lane);

    float m_run[8], l_run[8];
#pragma unroll
    for (int r = 0; r < 8; ++r) { m_run[r] = -1e30f; l_run[r] = 0.f; }

    int coll = lane & 15, half8 = (lane >> 4) * 8;

    auto process = [&](v8f acc, int colbase) {
        int col = colbase + coll;
#pragma unroll
        for (int r = 0; r < 8; ++r) {
            float s = acc[r] * scale;
            s = s + res * __sinf(s + ph);
            __builtin_nontemporal_store(s, &arow[(size_t)(q0 + r + half8) * Sq + col]);
            float mo = m_run[r];
            float mn = fmaxf(mo, s);
            l_run[r] = l_run[r] * __expf(mo - mn) + __expf(s - mn);
            m_run[r] = mn;
        }
    };

    // prologue: tile 0
    v8f acc_p;
    {
        v16bf b0 = load_b_kmajor(kbase, Sq, 0, 0, lane);
        v16bf b1 = load_b_kmajor(kbase, Sq, 32, 0, lane);
        v8f z = {};
        acc_p = wmma_bf16(a0, b0, z);
        acc_p = wmma_bf16(a1, b1, acc_p);
    }
    // steady state: WMMA(kt) overlaps elementwise processing of kt-16
    for (int kt = 16; kt < Sq; kt += 16) {
        v16bf b0 = load_b_kmajor(kbase, Sq, 0, kt, lane);
        v16bf b1 = load_b_kmajor(kbase, Sq, 32, kt, lane);
        if (kt + 16 < Sq) {
            __builtin_prefetch(kbase + (size_t)lane * Sq + kt + 16, 0, 1);
            __builtin_prefetch(kbase + (size_t)(32 + lane) * Sq + kt + 16, 0, 1);
        }
        v8f z = {};
        v8f acc_c = wmma_bf16(a0, b0, z);
        acc_c = wmma_bf16(a1, b1, acc_c);
        process(acc_p, kt - 16);
        acc_p = acc_c;
    }
    process(acc_p, Sq - 16);

#pragma unroll
    for (int r = 0; r < 8; ++r) {
        float m = m_run[r], l = l_run[r];
#pragma unroll
        for (int off = 1; off <= 8; off <<= 1) {
            float m2 = __shfl_xor(m, off, 16);
            float l2 = __shfl_xor(l, off, 16);
            float mn = fmaxf(m, m2);
            l = l * __expf(m - mn) + l2 * __expf(m2 - mn);
            m = mn;
        }
        if (coll == 0) {
            size_t o = (size_t)bh * Sq + q0 + r + half8;
            Mrow[o] = m;
            Lrow[o] = l;
        }
    }
}

// ---------------- Kernel 3: normalize + P @ V with TDM-streamed V tiles
__global__ void pv_kernel(const __bf16* __restrict__ vb,
                          const float* __restrict__ Mrow,
                          const float* __restrict__ Lrow,
                          float* __restrict__ attn,
                          __bf16* __restrict__ Ob) {
    __shared__ __bf16 Pb[2][8][16][32];   // per-wave P tiles, double buffered (16KB)
    __shared__ __bf16 Vt[3][32][64];      // block-shared TDM V tiles (12KB)
    __shared__ __bf16 Ot[8][16][64];      // output staging (16KB)
    const int lane = threadIdx.x & 31, wave = threadIdx.x >> 5;
    const int qtiles = Sq / 128;
    int bh = blockIdx.x / qtiles;
    int q0 = (blockIdx.x % qtiles) * 128 + wave * 16;

    const __bf16* vbase = vb + (size_t)bh * Sq * Hd;
    float* arow = attn + (size_t)bh * Sq * Sq;
    int coll = lane & 15, half8 = (lane >> 4) * 8;

    int nrow = lane >> 1, ncb = (lane & 1) * 16;
    float Mr  = Mrow[(size_t)bh * Sq + q0 + nrow];
    float Lir = 1.0f / Lrow[(size_t)bh * Sq + q0 + nrow];

    auto normalize = [&](int kt, int pb) {
        float* rp = arow + (size_t)(q0 + nrow) * Sq + kt + ncb;
        v16bf pvv;
#pragma unroll
        for (int u = 0; u < 4; ++u) {
            v4f s = __builtin_nontemporal_load((const v4f*)(rp + u * 4));
            v4f p;
#pragma unroll
            for (int j = 0; j < 4; ++j) {
                p[j] = __expf(s[j] - Mr) * Lir;
                pvv[u * 4 + j] = (__bf16)p[j];
            }
            __builtin_nontemporal_store(p, (v4f*)(rp + u * 4));
        }
        *(v16bf*)&Pb[pb][wave][nrow][ncb] = pvv;
    };

    // prologue: TDM tile 0, normalize tile 0
    if (wave == 0) {
        tdm_load_tile64(vbase, (unsigned int)(uintptr_t)&Vt[0][0][0], 32);
        __builtin_amdgcn_s_wait_tensorcnt((short)0);
    }
    normalize(0, 0);
    __syncthreads();                       // V tile 0 ready, published

    v8f acc0 = {}, acc1 = {}, acc2 = {}, acc3 = {};
    for (int kt = 0; kt < Sq; kt += 32) {
        int it = kt >> 5, slot = it % 3, pb = it & 1;
        // issue next V tile (slot's previous reader finished 2 barriers ago)
        if (wave == 0 && kt + 32 < Sq)
            tdm_load_tile64(vbase + (size_t)(kt + 32) * Hd,
                            (unsigned int)(uintptr_t)&Vt[(it + 1) % 3][0][0], 32);

        __builtin_amdgcn_wave_barrier();
        asm volatile("s_wait_dscnt 0x0" ::: "memory");
        v16bf a;
        {
            int m = lane & 15, koff = (lane < 16) ? 0 : 8;
            v8bf lo = *(const v8bf*)&Pb[pb][wave][m][koff];
            v8bf hi = *(const v8bf*)&Pb[pb][wave][m][16 + koff];
#pragma unroll
            for (int j = 0; j < 8; ++j) { a[j] = lo[j]; a[8 + j] = hi[j]; }
        }
        // B fragments from the LDS V tile (lane = column, elements over key rows, table order)
        v16bf b0, b1, b2, b3;
        {
            int n = lane & 15, koff = (lane < 16) ? 0 : 8;
#pragma unroll
            for (int j = 0; j < 8; ++j) {
                int k1 = koff + j, k2 = 16 + koff + j;
                b0[j] = Vt[slot][k1][n];      b0[8 + j] = Vt[slot][k2][n];
                b1[j] = Vt[slot][k1][16 + n]; b1[8 + j] = Vt[slot][k2][16 + n];
                b2[j] = Vt[slot][k1][32 + n]; b2[8 + j] = Vt[slot][k2][32 + n];
                b3[j] = Vt[slot][k1][48 + n]; b3[8 + j] = Vt[slot][k2][48 + n];
            }
        }
        // overlap: normalize next P tile while WMMAs run
        if (kt + 32 < Sq) normalize(kt + 32, pb ^ 1);
        acc0 = wmma_bf16(a, b0, acc0);
        acc1 = wmma_bf16(a, b1, acc1);
        acc2 = wmma_bf16(a, b2, acc2);
        acc3 = wmma_bf16(a, b3, acc3);

        if (wave == 0 && kt + 32 < Sq)
            __builtin_amdgcn_s_wait_tensorcnt((short)0);   // next tile landed
        __syncthreads();                                   // publish to all waves
    }

    int b_ = bh >> 4, h = bh & (Hn - 1);
#pragma unroll
    for (int r = 0; r < 8; ++r) {
        Ot[wave][r + half8][coll]      = (__bf16)acc0[r];
        Ot[wave][r + half8][16 + coll] = (__bf16)acc1[r];
        Ot[wave][r + half8][32 + coll] = (__bf16)acc2[r];
        Ot[wave][r + half8][48 + coll] = (__bf16)acc3[r];
    }
    __builtin_amdgcn_wave_barrier();
    asm volatile("s_wait_dscnt 0x0" ::: "memory");
    int row = lane >> 1, off = (lane & 1) * 32;
    __bf16* dp = Ob + ((size_t)b_ * Sq + q0 + row) * Dm + h * Hd + off;
    *(v16bf*)dp        = *(const v16bf*)&Ot[wave][row][off];
    *(v16bf*)(dp + 16) = *(const v16bf*)&Ot[wave][row][off + 16];
}

// ---------------- Kernel 4: output projection  out = O @ Wo^T + bo  (fp32)
__global__ void proj_out_kernel(const __bf16* __restrict__ Ob,
                                const __bf16* __restrict__ Wt,
                                const float* __restrict__ bias,
                                float* __restrict__ Y) {
    const int lane = threadIdx.x & 31, wave = threadIdx.x >> 5;
    const int nchunks = Dm / 64;
    int t  = blockIdx.x * 8 + wave;
    int r0 = (t / nchunks) * 16;
    int n0 = (t % nchunks) * 64;

    v8f acc[4] = {};
    for (int k0 = 0; k0 < Dm; k0 += 32) {
        v16bf a = load_a_bf16v(Ob, Dm, r0, k0, lane);
#pragma unroll
        for (int c = 0; c < 4; ++c) {
            v16bf b = load_b_kmajor(Wt, Dm, k0, n0 + c * 16, lane);
            acc[c] = wmma_bf16(a, b, acc[c]);
        }
    }
    int half8 = (lane >> 4) * 8, coll = lane & 15;
#pragma unroll
    for (int c = 0; c < 4; ++c) {
        int col = n0 + c * 16 + coll;
        float bvv = bias[col];
#pragma unroll
        for (int r = 0; r < 8; ++r)
            Y[(size_t)(r0 + r + half8) * Dm + col] = acc[c][r] + bvv;
    }
}

extern "C" void kernel_launch(void* const* d_in, const int* in_sizes, int n_in,
                              void* d_out, int out_size, void* d_ws, size_t ws_size,
                              hipStream_t stream) {
    (void)in_sizes; (void)n_in; (void)out_size; (void)ws_size;
    const float* query = (const float*)d_in[0];
    const float* key   = (const float*)d_in[1];
    const float* value = (const float*)d_in[2];
    const float* Wq = (const float*)d_in[3];
    const float* bq = (const float*)d_in[4];
    const float* Wk = (const float*)d_in[5];
    const float* bk = (const float*)d_in[6];
    const float* Wv = (const float*)d_in[7];
    const float* bv = (const float*)d_in[8];
    const float* Wo = (const float*)d_in[9];
    const float* bo = (const float*)d_in[10];
    const float* resonance = (const float*)d_in[11];
    const float* phase     = (const float*)d_in[12];

    float* out  = (float*)d_out;
    float* attn = out + (size_t)Bc * Sq * Dm;

    const size_t QKV = (size_t)Bc * Hn * Sq * Hd;
    const size_t WSZ = (size_t)Dm * Dm;
    __bf16* qb  = (__bf16*)d_ws;
    __bf16* kbT = qb  + QKV;
    __bf16* vb  = kbT + QKV;
    __bf16* Ob  = vb  + QKV;
    __bf16* WtQ = Ob  + QKV;
    __bf16* WtK = WtQ + WSZ;
    __bf16* WtV = WtK + WSZ;
    __bf16* WtO = WtV + WSZ;
    float* Mrow = (float*)(WtO + WSZ);
    float* Lrow = Mrow + (size_t)Bc * Hn * Sq;

    dim3 blk(256);
    const int wt_blocks   = (Dm / 32) * (Dm / 32);
    const int proj_blocks = ((Bc * Sq) / 16) * (Dm / 64) / 8;
    const int attn_blocks = Bc * Hn * (Sq / 128);

    wt_kernel<<<wt_blocks, blk, 0, stream>>>(Wq, WtQ);
    wt_kernel<<<wt_blocks, blk, 0, stream>>>(Wk, WtK);
    wt_kernel<<<wt_blocks, blk, 0, stream>>>(Wv, WtV);
    wt_kernel<<<wt_blocks, blk, 0, stream>>>(Wo, WtO);

    proj_qkv_kernel<0><<<proj_blocks, blk, 0, stream>>>(query, WtQ, bq, qb);
    proj_qkv_kernel<1><<<proj_blocks, blk, 0, stream>>>(key,   WtK, bk, kbT);
    proj_qkv_kernel<0><<<proj_blocks, blk, 0, stream>>>(value, WtV, bv, vb);

    scores_kernel<<<attn_blocks, blk, 0, stream>>>(qb, kbT, resonance, phase,
                                                   attn, Mrow, Lrow);
    pv_kernel<<<attn_blocks, blk, 0, stream>>>(vb, Mrow, Lrow, attn, Ob);
    proj_out_kernel<<<attn_blocks * 0 + proj_blocks, blk, 0, stream>>>(Ob, WtO, bo, out);
}
